// MultiHeadAttentionBlock_13554916786380
// MI455X (gfx1250) — matrix operand order repaired and checked by reading, not compiled, not run on hardware
//
#include <hip/hip_runtime.h>
#include <hip/hip_bf16.h>

// MI455X / gfx1250 fused multi-head attention block.
// Compute-bound (~69 GFLOP vs ~3.4us of HBM traffic at 23.3 TB/s), so all
// GEMMs run through V_WMMA_F32_16X16X32_BF16 (16384 FLOP/instr, 8x the
// density of the f32 16x16x4 WMMA). bf16 operands, f32 accumulation,
// f32 online softmax. bf16 tiles are staged into LDS with the CDNA5 async
// paths: global_load_async_to_lds_b128 (ASYNCcnt) and the Tensor Data Mover
// (tensor_load_to_lds, TENSORcnt, with LDS pad fields for bank-conflict-free
// strides). Converting stages (f32->bf16) stay on the VALU path.

typedef __attribute__((ext_vector_type(16))) __bf16          v16bf;
typedef __attribute__((ext_vector_type(8)))  float           v8f;
typedef __attribute__((ext_vector_type(4)))  unsigned short  ushortx4;
typedef __attribute__((ext_vector_type(4)))  unsigned int    uint32x4;
typedef __attribute__((ext_vector_type(8)))  int             int32x8;
typedef __attribute__((ext_vector_type(4)))  int             int32x4;

#define Bb  2
#define Ss  2048
#define Dd  1024
#define Hh  16
#define DKk 64

#if __has_builtin(__builtin_amdgcn_tensor_load_to_lds) && \
    __has_builtin(__builtin_amdgcn_s_wait_tensorcnt)
#define HAVE_TDM 1
#endif

union FragU { uint4 u4[2]; v16bf v; };

__device__ __forceinline__ unsigned short to_bf16(float f) {
  unsigned int u = __float_as_uint(f);
  u += 0x7fffu + ((u >> 16) & 1u);          // round-to-nearest-even
  return (unsigned short)(u >> 16);
}

__device__ __forceinline__ v8f wmma_bf16(v16bf a, v16bf b, v8f c) {
  return __builtin_amdgcn_wmma_f32_16x16x32_bf16(false, a, false, b,
                                                 (short)0, c, false, false);
}

__device__ __forceinline__ unsigned lds_off(const void* p) {
  // generic pointers to LDS carry the LDS offset in the low 32 bits
  return (unsigned)(unsigned long long)p;
}

// async global->LDS 16-byte copy per lane; completion via ASYNCcnt
__device__ __forceinline__ void async_copy_b128(unsigned lofs, const void* g) {
  asm volatile("global_load_async_to_lds_b128 %0, %1, off"
               :: "v"(lofs), "v"((unsigned long long)g) : "memory");
}
__device__ __forceinline__ void wait_asynccnt0() {
  asm volatile("s_wait_asynccnt 0x0" ::: "memory");
}

// A fragment: 16x32 bf16 tile stored [row][K] in LDS (stride in halfwords).
// lane<16: M=lane, K = {0..7, 16..23}; lane>=16: M=lane-16, K = {8..15, 24..31}.
__device__ __forceinline__ v16bf load_frag_a(const unsigned short* lds, int stride) {
  const int lane = threadIdx.x & 31;
  const int m    = lane & 15;
  const int koff = (lane >> 4) << 3;        // 0 or 8
  const unsigned short* p = lds + m * stride + koff;
  FragU f;
  f.u4[0] = *(const uint4*)(p);
  f.u4[1] = *(const uint4*)(p + 16);
  return f.v;
}

// B fragment: 32x16 bf16, tile stored [N][K] in LDS (K contiguous).
// lane<16: N=lane, K = {0..7, 16..23}; lane>=16: N=lane-16, K = {8..15, 24..31}.
__device__ __forceinline__ v16bf load_frag_b(const unsigned short* lds, int stride) {
  const int lane = threadIdx.x & 31;
  const int n    = lane & 15;
  const int koff = (lane >> 4) << 3;        // 0 or 8
  const unsigned short* p = lds + n * stride + koff;
  FragU f;
  f.u4[0] = *(const uint4*)(p);
  f.u4[1] = *(const uint4*)(p + 16);
  return f.v;
}

__device__ __forceinline__ float half_red_max(float v) {
  for (int m = 1; m < 16; m <<= 1) v = fmaxf(v, __shfl_xor(v, m));
  return v;
}
__device__ __forceinline__ float half_red_sum(float v) {
  for (int m = 1; m < 16; m <<= 1) v += __shfl_xor(v, m);
  return v;
}

// ---------------------------------------------------------------------------
// Kernel 1: Q/K/V projections. C[m,n] = sum_k X[m,k] * W[n,k]  (y = x @ W^T)
// Output bf16 in [B,H,S,DK] layout. Block: 256 thr (8 waves), tile 128x64.
// f32->bf16 conversion happens while staging, so this stays on the VALU path.
// ---------------------------------------------------------------------------
__global__ __launch_bounds__(256) void qkv_proj(
    const float* __restrict__ q, const float* __restrict__ k,
    const float* __restrict__ v,
    const float* __restrict__ wq, const float* __restrict__ wk,
    const float* __restrict__ wv,
    unsigned short* __restrict__ Qo, unsigned short* __restrict__ Ko,
    unsigned short* __restrict__ Vo) {
  const float* X; const float* W; unsigned short* O;
  if (blockIdx.z == 0)      { X = q; W = wq; O = Qo; }
  else if (blockIdx.z == 1) { X = k; W = wk; O = Ko; }
  else                      { X = v; W = wv; O = Vo; }

  const int m0 = blockIdx.y * 128;
  const int n0 = blockIdx.x * 64;
  const int tid = threadIdx.x, wv_ = tid >> 5, lane = tid & 31;

  __shared__ unsigned short Xs[128 * 40];   // [row][k] k-step=32, pad->40
  __shared__ unsigned short Wts[64 * 40];

  v8f acc[4] = {};

  for (int k0 = 0; k0 < Dd; k0 += 32) {
    __syncthreads();
    for (int i = 0; i < 4; ++i) {                    // X tile 128x32 f32->bf16
      int e = tid + i * 256, row = e >> 3, c4 = (e & 7) << 2;
      float4 f = *(const float4*)(X + (size_t)(m0 + row) * Dd + k0 + c4);
      ushortx4 hv = {to_bf16(f.x), to_bf16(f.y), to_bf16(f.z), to_bf16(f.w)};
      *(ushortx4*)(&Xs[row * 40 + c4]) = hv;
    }
    for (int i = 0; i < 2; ++i) {                    // W tile 64x32
      int e = tid + i * 256, row = e >> 3, c4 = (e & 7) << 2;
      float4 f = *(const float4*)(W + (size_t)(n0 + row) * Dd + k0 + c4);
      ushortx4 hv = {to_bf16(f.x), to_bf16(f.y), to_bf16(f.z), to_bf16(f.w)};
      *(ushortx4*)(&Wts[row * 40 + c4]) = hv;
    }
    __syncthreads();
    v16bf a = load_frag_a(Xs + wv_ * 16 * 40, 40);
    for (int ns = 0; ns < 4; ++ns) {
      v16bf bf = load_frag_b(Wts + ns * 16 * 40, 40);
      acc[ns] = wmma_bf16(a, bf, acc[ns]);
    }
  }

  const int hi8 = (lane >> 4) << 3, cn = lane & 15;
  for (int ns = 0; ns < 4; ++ns)
    for (int r = 0; r < 8; ++r) {
      int m = m0 + wv_ * 16 + r + hi8;
      int n = n0 + ns * 16 + cn;
      int b = m >> 11, s = m & (Ss - 1), h = n >> 6, dk = n & (DKk - 1);
      O[((size_t)(b * Hh + h) * Ss + s) * DKk + dk] = to_bf16(acc[ns][r]);
    }
}

// ---------------------------------------------------------------------------
// Kernel 2: flash attention with online softmax. Each wave owns a 16-row Q
// tile (async-copied to LDS); block stages 32-row K tiles via the Tensor Data
// Mover (padded LDS rows) and V tiles transposed via VALU. scores = QK^T/8 +
// mask, P = softmax, O = P V. Output bf16 [B,H,S,DK].
// ---------------------------------------------------------------------------
__global__ __launch_bounds__(256) void flash_attn(
    const unsigned short* __restrict__ Q, const unsigned short* __restrict__ K,
    const unsigned short* __restrict__ V, const int* __restrict__ mask,
    unsigned short* __restrict__ Oo) {
  const int tid = threadIdx.x, w = tid >> 5, lane = tid & 31;
  const int b = blockIdx.z, h = blockIdx.y;
  const int q0 = blockIdx.x * 128 + w * 16;
  const size_t bh = (size_t)(b * Hh + h) * Ss;
  const int hi8 = (lane >> 4) << 3, cn = lane & 15;

  __shared__ unsigned short Qs[8][16 * 72];   // per-wave Q tile 16x64
  __shared__ unsigned short Ks[32 * 72];      // K tile [kj][dk], stride 72
  __shared__ unsigned short Vt[64 * 40];      // V tile transposed [dk][kj]
  __shared__ unsigned short Ps[8][16 * 40];   // per-wave P tile 16x32
  __shared__ float mAdd[32];

  {  // stage own Q tile: async global->LDS (ASYNCcnt), no VGPR round-trip
    const unsigned qlds = lds_off(&Qs[w][0]);
    for (int i = 0; i < 4; ++i) {
      int e = lane + 32 * i, row = e >> 3, seg = e & 7;
      async_copy_b128(qlds + (unsigned)((row * 72 + seg * 8) * 2),
                      &Q[(bh + q0 + row) * DKk + seg * 8]);
    }
    wait_asynccnt0();
  }

  v8f o0 = {}, o1 = {}, o2 = {}, o3 = {};
  v8f mrow, lrow;
  for (int r = 0; r < 8; ++r) { mrow[r] = -1e30f; lrow[r] = 0.0f; }

  for (int kv0 = 0; kv0 < Ss; kv0 += 32) {
    __syncthreads();
#ifdef HAVE_TDM
    if (w == 0) {   // K tile 32x64 bf16 via TDM, one wave issues the descriptor
      const unsigned klds = lds_off(&Ks[0]);
      unsigned long long ga = (unsigned long long)&K[(bh + kv0) * DKk];
      // D# group0: count=1, lds_addr, global_addr[56:0], type=2
      uint32x4 g0 = { 1u, klds, (unsigned)ga,
                      (unsigned)((ga >> 32) & 0x01FFFFFFull) | 0x80000000u };
      // D# group1: data_size=2B (code 1), pad_enable, pad_interval=32 DW
      // (code 4: one 128B row), pad_amount=4 DW (code 3: 16B -> stride 72 hw),
      // tensor_dim0=64, tensor_dim1=32, tile_dim0=64, tile_dim1=32,
      // tensor_dim0_stride=64
      int32x8 g1 = { (int)((1u << 16) | (1u << 20) | (4u << 22) | (3u << 25)),
                     (int)(64u << 16), (int)(32u << 16), (int)(64u << 16),
                     32, 64, 0, 0 };
      int32x4 z4 = {0, 0, 0, 0};
#if __clang_major__ >= 23
      int32x8 z8 = {0, 0, 0, 0, 0, 0, 0, 0};
      __builtin_amdgcn_tensor_load_to_lds(g0, g1, z4, z4, z8, 0);
#else
      __builtin_amdgcn_tensor_load_to_lds(g0, g1, z4, z4, 0);
#endif
      __builtin_amdgcn_s_wait_tensorcnt(0);
    }
#else
    {  // K tile via async global->LDS copies, all waves participate
      const unsigned klds = lds_off(&Ks[0]);
      int row = tid >> 3, seg = tid & 7;
      async_copy_b128(klds + (unsigned)((row * 72 + seg * 8) * 2),
                      &K[(bh + kv0 + row) * DKk + seg * 8]);
      wait_asynccnt0();
    }
#endif
    for (int i = 0; i < 8; ++i) {             // V tile, transposed into LDS
      int e = tid + 256 * i, kj = e >> 6, dk = e & 63;
      Vt[dk * 40 + kj] = V[(bh + kv0 + kj) * DKk + dk];
    }
    if (tid < 32) mAdd[tid] = (mask[b * Ss + kv0 + tid] == 0) ? -1e9f : 0.0f;
    __syncthreads();

    // scores: two 16x16 C tiles, K-dim 64 = 2 WMMA steps each
    v16bf a0 = load_frag_a(Qs[w], 72);
    v16bf a1 = load_frag_a(Qs[w] + 32, 72);
    v8f s0 = {}, s1 = {};
    s0 = wmma_bf16(a0, load_frag_b(Ks, 72), s0);
    s0 = wmma_bf16(a1, load_frag_b(Ks + 32, 72), s0);
    s1 = wmma_bf16(a0, load_frag_b(Ks + 16 * 72, 72), s1);
    s1 = wmma_bf16(a1, load_frag_b(Ks + 16 * 72 + 32, 72), s1);

    float madd0 = mAdd[cn], madd1 = mAdd[16 + cn];
    v8f rmax;
    for (int r = 0; r < 8; ++r) {
      s0[r] = s0[r] * 0.125f + madd0;         // 1/sqrt(DK) = 1/8
      s1[r] = s1[r] * 0.125f + madd1;
      rmax[r] = half_red_max(fmaxf(s0[r], s1[r]));
    }
    v8f corr;
    for (int r = 0; r < 8; ++r) {             // online softmax update
      float mn = fmaxf(mrow[r], rmax[r]);
      float c  = __expf(mrow[r] - mn);
      mrow[r] = mn; corr[r] = c;
      s0[r] = __expf(s0[r] - mn);
      s1[r] = __expf(s1[r] - mn);
      lrow[r] = lrow[r] * c + half_red_sum(s0[r] + s1[r]);
    }
    for (int r = 0; r < 8; ++r) {
      o0[r] *= corr[r]; o1[r] *= corr[r]; o2[r] *= corr[r]; o3[r] *= corr[r];
    }

    unsigned short* Pw = Ps[w];               // C-layout -> A-layout via LDS
    for (int r = 0; r < 8; ++r) {
      int row = r + hi8;
      Pw[row * 40 + cn]      = to_bf16(s0[r]);
      Pw[row * 40 + 16 + cn] = to_bf16(s1[r]);
    }
    asm volatile("s_wait_dscnt 0x0" ::: "memory");
    v16bf pa = load_frag_a(Pw, 40);
    o0 = wmma_bf16(pa, load_frag_b(Vt,           40), o0);
    o1 = wmma_bf16(pa, load_frag_b(Vt + 16 * 40, 40), o1);
    o2 = wmma_bf16(pa, load_frag_b(Vt + 32 * 40, 40), o2);
    o3 = wmma_bf16(pa, load_frag_b(Vt + 48 * 40, 40), o3);
  }

  for (int r = 0; r < 8; ++r) {
    float inv = 1.0f / lrow[r];
    size_t base = (bh + q0 + r + hi8) * DKk;
    Oo[base + cn]      = to_bf16(o0[r] * inv);
    Oo[base + 16 + cn] = to_bf16(o1[r] * inv);
    Oo[base + 32 + cn] = to_bf16(o2[r] * inv);
    Oo[base + 48 + cn] = to_bf16(o3[r] * inv);
  }
}

// ---------------------------------------------------------------------------
// Kernel 3: output projection. A (bf16, concat-heads gathered from
// [B,H,S,DK], async-copied to LDS) @ w_o^T -> f32 d_out [B,S,D].
// ---------------------------------------------------------------------------
__global__ __launch_bounds__(256) void out_proj(
    const unsigned short* __restrict__ A, const float* __restrict__ Wo,
    float* __restrict__ Y) {
  const int m0 = blockIdx.y * 128, n0 = blockIdx.x * 64;
  const int tid = threadIdx.x, w = tid >> 5, lane = tid & 31;

  __shared__ unsigned short Xs[128 * 40];
  __shared__ unsigned short Wts[64 * 40];

  const unsigned xlds = lds_off(&Xs[0]);
  v8f acc[4] = {};

  for (int k0 = 0; k0 < Dd; k0 += 32) {
    __syncthreads();
    const int hh = k0 >> 6, dk0 = k0 & (DKk - 1); // concat-heads gather
    for (int i = 0; i < 2; ++i) {                 // A tile: async global->LDS
      int e = tid + i * 256, row = e >> 2, seg = e & 3;
      int m = m0 + row, b = m >> 11, s = m & (Ss - 1);
      async_copy_b128(xlds + (unsigned)((row * 40 + seg * 8) * 2),
                      &A[((size_t)(b * Hh + hh) * Ss + s) * DKk + dk0 + seg * 8]);
    }
    for (int i = 0; i < 2; ++i) {                 // W tile: f32->bf16 via VALU
      int e = tid + i * 256, row = e >> 3, c4 = (e & 7) << 2;
      float4 f = *(const float4*)(Wo + (size_t)(n0 + row) * Dd + k0 + c4);
      ushortx4 hv = {to_bf16(f.x), to_bf16(f.y), to_bf16(f.z), to_bf16(f.w)};
      *(ushortx4*)(&Wts[row * 40 + c4]) = hv;
    }
    wait_asynccnt0();
    __syncthreads();
    v16bf a = load_frag_a(Xs + w * 16 * 40, 40);
    for (int ns = 0; ns < 4; ++ns) {
      v16bf bf = load_frag_b(Wts + ns * 16 * 40, 40);
      acc[ns] = wmma_bf16(a, bf, acc[ns]);
    }
  }

  const int hi8 = (lane >> 4) << 3, cn = lane & 15;
  for (int ns = 0; ns < 4; ++ns)
    for (int r = 0; r < 8; ++r) {
      int m = m0 + w * 16 + r + hi8;
      int n = n0 + ns * 16 + cn;
      Y[(size_t)m * Dd + n] = acc[ns][r];
    }
}

extern "C" void kernel_launch(void* const* d_in, const int* in_sizes, int n_in,
                              void* d_out, int out_size, void* d_ws, size_t ws_size,
                              hipStream_t stream) {
  (void)in_sizes; (void)n_in; (void)out_size; (void)ws_size;
  const float* q    = (const float*)d_in[0];
  const float* k    = (const float*)d_in[1];
  const float* v    = (const float*)d_in[2];
  const int*   mask = (const int*)d_in[3];
  const float* wq   = (const float*)d_in[4];
  const float* wk   = (const float*)d_in[5];
  const float* wv   = (const float*)d_in[6];
  const float* wo   = (const float*)d_in[7];
  float* out = (float*)d_out;

  const size_t per = (size_t)Bb * Hh * Ss * DKk;   // 4M bf16 elems = 8 MB
  unsigned short* Qws = (unsigned short*)d_ws;
  unsigned short* Kws = Qws + per;
  unsigned short* Vws = Kws + per;
  unsigned short* Ows = Vws + per;

  // 1) Q/K/V projections: M=4096, N=1024, tiles 128x64, z selects matrix
  qkv_proj<<<dim3(Dd / 64, (Bb * Ss) / 128, 3), 256, 0, stream>>>(
      q, k, v, wq, wk, wv, Qws, Kws, Vws);
  // 2) flash attention: 16 q-tiles x 16 heads x 2 batches
  flash_attn<<<dim3(Ss / 128, Hh, Bb), 256, 0, stream>>>(
      Qws, Kws, Vws, mask, Ows);
  // 3) output projection
  out_proj<<<dim3(Dd / 64, (Bb * Ss) / 128, 1), 256, 0, stream>>>(
      Ows, wo, out);
}
